// ELA_65231963291734
// MI455X (gfx1250) — matrix operand order repaired and checked by reading, not compiled, not run on hardware
//
#include <hip/hip_runtime.h>
#include <hip/hip_bf16.h>

typedef __bf16 bf16_t;
typedef __attribute__((ext_vector_type(16))) __bf16 v16bf;
typedef __attribute__((ext_vector_type(8)))  __bf16 v8bf;
typedef __attribute__((ext_vector_type(8)))  float  v8f;

#define B_    16
#define DIM_  512
#define NPIX  1024     // 32*32
#define DQK   512
#define DV    1024
#define ATTN  2048     // 2*DQK + DV
#define SCALE_ 0.03125f // n^-0.5 = 1/32
#define EPS_  1e-6f

// ---------------------------------------------------------------------------
// Batched WMMA GEMM:  C[M,N] = A[M,K] * Bt[N,K]^T   (bf16 in, f32 out)
// Block = 256 threads (8 waves). Block tile 64(M) x 128(N), K-step 32.
// CDNA5 async staging: global_load_async_to_lds_b128 double-buffers the
// 64x32 A tile and 128x32 B tile in LDS (ASYNCcnt), WMMAs read via ds_load.
// Each wave: 32x32 output = 2x2 v_wmma_f32_16x16x32_bf16, fp32 accumulate.
// TRANS_OUT=true stores C^T (N x M, ldc=M) with contiguous 8-float runs.
// ---------------------------------------------------------------------------
template<bool TRANS_OUT>
__global__ __launch_bounds__(256)
void wmma_gemm_bf16(const bf16_t* __restrict__ A,  long aBatch, int lda,
                    const bf16_t* __restrict__ Bt, long bBatch, int ldb,
                    float* __restrict__ C,         long cBatch, int ldc,
                    const float* __restrict__ bias, int K)
{
    __shared__ bf16_t As[2][64 * 32];    // [buf][row*32 + k]
    __shared__ bf16_t Bs[2][128 * 32];

    const int bz = blockIdx.z;
    A  += (long)bz * aBatch;
    Bt += (long)bz * bBatch;
    C  += (long)bz * cBatch;

    const int tid  = threadIdx.x;
    const int lane = tid & 31;
    const int wave = tid >> 5;

    // --- cooperative async staging addressing ---
    // lane -> (row within 8-row group, 16B chunk within 64B row)
    const int srow = lane >> 2;          // 0..7
    const int scol = (lane & 3) * 8;     // element: 0,8,16,24 (16B chunks)
    const int arow  = wave * 8  + srow;  // A: 64 rows across 8 waves
    const int brow0 = wave * 16 + srow;  // B: 128 rows, 2 issues per wave
    const int brow1 = brow0 + 8;

    const int blkM = blockIdx.y * 64;
    const int blkN = blockIdx.x * 128;

    const long aOff  = (long)(blkM + arow)  * lda + scol;  // element offsets
    const long bOff0 = (long)(blkN + brow0) * ldb + scol;
    const long bOff1 = (long)(blkN + brow1) * ldb + scol;

    const unsigned aLds  = (unsigned)(arow  * 32 + scol) * 2u;  // byte offsets
    const unsigned bLds0 = (unsigned)(brow0 * 32 + scol) * 2u;
    const unsigned bLds1 = (unsigned)(brow1 * 32 + scol) * 2u;

    auto stage = [&](int buf, int k0) {
        const unsigned asB = (unsigned)(size_t)&As[buf][0];
        const unsigned bsB = (unsigned)(size_t)&Bs[buf][0];
        const int aByte  = (int)((aOff  + k0) * 2);
        const int bByte0 = (int)((bOff0 + k0) * 2);
        const int bByte1 = (int)((bOff1 + k0) * 2);
        // GVS mode: mem = SADDR(64b) + VADDR(32b) ; LDS dst = LDS_BASE + VDST
        asm volatile("global_load_async_to_lds_b128 %0, %1, %2"
                     :: "v"(asB + aLds),  "v"(aByte),  "s"(A)  : "memory");
        asm volatile("global_load_async_to_lds_b128 %0, %1, %2"
                     :: "v"(bsB + bLds0), "v"(bByte0), "s"(Bt) : "memory");
        asm volatile("global_load_async_to_lds_b128 %0, %1, %2"
                     :: "v"(bsB + bLds1), "v"(bByte1), "s"(Bt) : "memory");
    };

    // --- per-wave WMMA fragment addressing (reads from LDS tiles) ---
    const int wm   = wave & 1;           // 2 waves along M
    const int wn   = wave >> 1;          // 4 waves along N
    const int m0   = blkM + wm * 32;
    const int n0   = blkN + wn * 32;
    const int lrow = lane & 15;
    const int lhi  = lane >> 4;          // 0 or 1

    const int am0 = (wm * 32 + lrow) * 32;        // local A row offsets (elems)
    const int am1 = (wm * 32 + 16 + lrow) * 32;
    const int bn0 = (wn * 32 + lrow) * 32;        // local B row offsets
    const int bn1 = (wn * 32 + 16 + lrow) * 32;

    v8f acc00 = {}, acc01 = {}, acc10 = {}, acc11 = {};

    stage(0, 0);                                   // prologue: tile 0

    for (int k0 = 0, it = 0; k0 < K; k0 += 32, ++it) {
        const int buf = it & 1;
        // own async writes for tile `it` have landed:
        asm volatile("s_wait_asynccnt 0x0" ::: "memory");
        __syncthreads();                           // everyone's writes visible
        if (k0 + 32 < K) stage(buf ^ 1, k0 + 32);  // overlap next tile DMA

        const bf16_t* ap = &As[buf][0];
        const bf16_t* bp = &Bs[buf][0];

        v16bf a0, a1, b0, b1;
        // A 16x32 frag: lane(l<16): K {lhi*8..+7},{16+lhi*8..+7}
        ((v8bf*)&a0)[0] = *(const v8bf*)(ap + am0 + lhi * 8);
        ((v8bf*)&a0)[1] = *(const v8bf*)(ap + am0 + 16 + lhi * 8);
        ((v8bf*)&a1)[0] = *(const v8bf*)(ap + am1 + lhi * 8);
        ((v8bf*)&a1)[1] = *(const v8bf*)(ap + am1 + 16 + lhi * 8);
        // B 32x16 frag: lane(l<16): K {lhi*8..+7},{16+lhi*8..+7}
        ((v8bf*)&b0)[0] = *(const v8bf*)(bp + bn0 + lhi * 8);
        ((v8bf*)&b0)[1] = *(const v8bf*)(bp + bn0 + 16 + lhi * 8);
        ((v8bf*)&b1)[0] = *(const v8bf*)(bp + bn1 + lhi * 8);
        ((v8bf*)&b1)[1] = *(const v8bf*)(bp + bn1 + 16 + lhi * 8);

        acc00 = __builtin_amdgcn_wmma_f32_16x16x32_bf16(false, a0, false, b0, (short)0, acc00, false, false);
        acc01 = __builtin_amdgcn_wmma_f32_16x16x32_bf16(false, a0, false, b1, (short)0, acc01, false, false);
        acc10 = __builtin_amdgcn_wmma_f32_16x16x32_bf16(false, a1, false, b0, (short)0, acc10, false, false);
        acc11 = __builtin_amdgcn_wmma_f32_16x16x32_bf16(false, a1, false, b1, (short)0, acc11, false, false);

        __syncthreads();                 // readers done before buf is restaged
    }

    auto store_tile = [&](const v8f& acc, int mt, int nt) {
        const int mbase = m0 + mt * 16 + lhi * 8;  // 8 consecutive rows
        const int col   = n0 + nt * 16 + lrow;
        if constexpr (TRANS_OUT) {
            float* p = C + (long)col * ldc + mbase;
            #pragma unroll
            for (int e = 0; e < 8; ++e) {
                float v = acc[e];
                if (bias) v += bias[mbase + e];
                p[e] = v;
            }
        } else {
            #pragma unroll
            for (int e = 0; e < 8; ++e) {
                float v = acc[e];
                if (bias) v += bias[mbase + e];
                C[(long)(mbase + e) * ldc + col] = v;
            }
        }
    };
    store_tile(acc00, 0, 0); store_tile(acc01, 0, 1);
    store_tile(acc10, 1, 0); store_tile(acc11, 1, 1);
}

// ---------------------------------------------------------------------------
// Batched elementwise fp32 -> bf16 convert with optional relu and scale.
// ---------------------------------------------------------------------------
template<bool RELU>
__global__ void convert_bf16_k(const float* __restrict__ src, long srcBatch,
                               bf16_t* __restrict__ dst, long dstBatch,
                               long count, float scale)
{
    src += (long)blockIdx.y * srcBatch;
    dst += (long)blockIdx.y * dstBatch;
    for (long i = (long)blockIdx.x * blockDim.x + threadIdx.x; i < count;
         i += (long)gridDim.x * blockDim.x) {
        float v = src[i];
        if (RELU) v = fmaxf(v, 0.f);
        dst[i] = (bf16_t)(v * scale);
    }
}

// ---------------------------------------------------------------------------
// Tiled transpose + convert: src (R x C fp32, row-major) -> dst (C x R OutT).
// Grid: (C/32, R/32, batch). Block 256.
// ---------------------------------------------------------------------------
template<typename OutT, bool RELU>
__global__ __launch_bounds__(256)
void transpose_convert_k(const float* __restrict__ src, long srcBatch, int srcLd,
                         OutT* __restrict__ dst, long dstBatch, int dstLd,
                         float scale)
{
    __shared__ float tile[32][33];
    src += (long)blockIdx.z * srcBatch;
    dst += (long)blockIdx.z * dstBatch;
    const int r0 = blockIdx.y * 32;
    const int c0 = blockIdx.x * 32;
    const int tx = threadIdx.x & 31;
    const int ty = threadIdx.x >> 5;
    #pragma unroll
    for (int i = 0; i < 4; ++i) {
        int r = ty + i * 8;
        tile[r][tx] = src[(long)(r0 + r) * srcLd + c0 + tx];
    }
    __syncthreads();
    #pragma unroll
    for (int i = 0; i < 4; ++i) {
        int cc = ty + i * 8;
        float v = tile[tx][cc];
        if (RELU) v = fmaxf(v, 0.f);
        dst[(long)(c0 + cc) * dstLd + r0 + tx] = (OutT)(v * scale);
    }
}

// ---------------------------------------------------------------------------
// Depthwise 3x3 'SAME' conv + residual + bias: xf = y + dwconv(y) + b.
// One block per (channel, batch) 32x32 plane, haloed LDS tile.
// ---------------------------------------------------------------------------
__global__ __launch_bounds__(256)
void dwconv_k(const float* __restrict__ y, const float* __restrict__ w,
              const float* __restrict__ bias, float* __restrict__ xf)
{
    const int c = blockIdx.x, b = blockIdx.y;
    const float* yp = y  + ((long)b * ATTN + c) * NPIX;
    float*       xp = xf + ((long)b * ATTN + c) * NPIX;
    __shared__ float t[34 * 34];
    for (int i = threadIdx.x; i < 34 * 34; i += 256) {
        int hh = i / 34 - 1, ww = i % 34 - 1;
        t[i] = (hh >= 0 && hh < 32 && ww >= 0 && ww < 32) ? yp[hh * 32 + ww] : 0.f;
    }
    __syncthreads();
    float w9[9];
    #pragma unroll
    for (int j = 0; j < 9; ++j) w9[j] = w[c * 9 + j];
    const float bb = bias[c];
    for (int i = threadIdx.x; i < NPIX; i += 256) {
        int hh = i >> 5, ww = i & 31;
        float s = 0.f;
        #pragma unroll
        for (int kh = 0; kh < 3; ++kh)
            #pragma unroll
            for (int kw = 0; kw < 3; ++kw)
                s += w9[kh * 3 + kw] * t[(hh + kh) * 34 + (ww + kw)];
        xp[i] = yp[i] + s + bb;
    }
}

// kf_mean[b,d] = mean_n relu(xf[b, DQK+d, n])
__global__ __launch_bounds__(256)
void kmean_k(const float* __restrict__ xf, float* __restrict__ kf_mean)
{
    const int d = blockIdx.x, b = blockIdx.y;
    const float* kp = xf + ((long)b * ATTN + DQK + d) * NPIX;
    float s = 0.f;
    for (int i = threadIdx.x; i < NPIX; i += 256) s += fmaxf(kp[i], 0.f);
    __shared__ float red[256];
    red[threadIdx.x] = s;
    __syncthreads();
    for (int off = 128; off > 0; off >>= 1) {
        if (threadIdx.x < off) red[threadIdx.x] += red[threadIdx.x + off];
        __syncthreads();
    }
    if (threadIdx.x == 0) kf_mean[(long)b * DQK + d] = red[0] * (1.f / (float)NPIX);
}

// z[b,n] = dot(kf_mean[b,:], relu(q[b,:,n])) + EPS   (coalesced across n)
__global__ __launch_bounds__(256)
void z_k(const float* __restrict__ xf, const float* __restrict__ kf_mean,
         float* __restrict__ z)
{
    const int n = blockIdx.x * 256 + threadIdx.x;
    const int b = blockIdx.y;
    const float* qp = xf + (long)b * ATTN * NPIX;      // rows 0..DQK-1 are q
    const float* km = kf_mean + (long)b * DQK;
    float acc = 0.f;
    for (int d = 0; d < DQK; ++d)
        acc += km[d] * fmaxf(qp[(long)d * NPIX + n], 0.f);
    z[(long)b * NPIX + n] = acc + EPS_;
}

// Per-(b,n) column: a = (attn/z)*sigmoid(gate); RMS over DV channels;
// aT[n,v] = bf16(a/rms*norm_w + v_raw*scale_w)   (row-major for GEMM5 Bt)
__global__ __launch_bounds__(256)
void epilogue_k(const float* __restrict__ attnT, const float* __restrict__ gateT,
                const float* __restrict__ vT, const float* __restrict__ z,
                const float* __restrict__ norm_w, const float* __restrict__ scale_w,
                bf16_t* __restrict__ aT)
{
    const int n = blockIdx.x, b = blockIdx.y;
    const long base = ((long)b * NPIX + n) * DV;
    const float zinv = 1.f / z[(long)b * NPIX + n];
    float av[4], vv[4];
    float ss = 0.f;
    #pragma unroll
    for (int t = 0; t < 4; ++t) {
        int v = threadIdx.x + t * 256;
        float at = attnT[base + v] * zinv;
        float g  = 1.f / (1.f + __expf(-gateT[base + v]));
        float a  = at * g;
        av[t] = a; vv[t] = vT[base + v];
        ss += a * a;
    }
    __shared__ float red[256];
    red[threadIdx.x] = ss;
    __syncthreads();
    for (int off = 128; off > 0; off >>= 1) {
        if (threadIdx.x < off) red[threadIdx.x] += red[threadIdx.x + off];
        __syncthreads();
    }
    const float rinv = rsqrtf(red[0] * (1.f / (float)DV) + EPS_);
    #pragma unroll
    for (int t = 0; t < 4; ++t) {
        int v = threadIdx.x + t * 256;
        aT[base + v] = (bf16_t)(av[t] * rinv * norm_w[v] + vv[t] * scale_w[v]);
    }
}

// ---------------------------------------------------------------------------
extern "C" void kernel_launch(void* const* d_in, const int* in_sizes, int n_in,
                              void* d_out, int out_size, void* d_ws, size_t ws_size,
                              hipStream_t stream)
{
    const float* x       = (const float*)d_in[0];
    const float* proj_w  = (const float*)d_in[1];
    const float* proj_b  = (const float*)d_in[2];
    const float* dwc_w   = (const float*)d_in[3];
    const float* dwc_b   = (const float*)d_in[4];
    const float* g_w     = (const float*)d_in[5];
    const float* g_b     = (const float*)d_in[6];
    const float* norm_w  = (const float*)d_in[7];
    const float* scale_w = (const float*)d_in[8];
    const float* out_w   = (const float*)d_in[9];
    const float* out_b   = (const float*)d_in[10];
    float* out = (float*)d_out;

    char* ws = (char*)d_ws;
    size_t off = 0;
    auto take = [&](size_t bytes) -> void* {
        void* p = ws + off;
        off = (off + bytes + 255) & ~(size_t)255;
        return p;
    };

    bf16_t* proj_wb = (bf16_t*)take((size_t)ATTN * DIM_ * 2);
    bf16_t* g_wb    = (bf16_t*)take((size_t)DV * ATTN * 2);
    bf16_t* out_wb  = (bf16_t*)take((size_t)DIM_ * DV * 2);
    bf16_t* xT      = (bf16_t*)take((size_t)B_ * NPIX * DIM_ * 2);
    float*  y       = (float*)take((size_t)B_ * ATTN * NPIX * 4);
    float*  xf      = (float*)take((size_t)B_ * ATTN * NPIX * 4);
    bf16_t* xfT     = (bf16_t*)take((size_t)B_ * NPIX * ATTN * 2);
    bf16_t* qfT     = (bf16_t*)take((size_t)B_ * NPIX * DQK * 2);
    bf16_t* kf_s    = (bf16_t*)take((size_t)B_ * DQK * NPIX * 2);
    bf16_t* v_s     = (bf16_t*)take((size_t)B_ * DV * NPIX * 2);
    float*  vT      = (float*)take((size_t)B_ * NPIX * DV * 4);
    float*  vkT     = (float*)take((size_t)B_ * DV * DQK * 4);
    bf16_t* vkTb    = (bf16_t*)take((size_t)B_ * DV * DQK * 2);
    float*  attnT   = (float*)take((size_t)B_ * NPIX * DV * 4);
    float*  gateT   = (float*)take((size_t)B_ * NPIX * DV * 4);
    float*  kf_mean = (float*)take((size_t)B_ * DQK * 4);
    float*  zbuf    = (float*)take((size_t)B_ * NPIX * 4);
    bf16_t* aT      = (bf16_t*)take((size_t)B_ * NPIX * DV * 2);

    // Weights -> bf16 (batch-independent)
    convert_bf16_k<false><<<dim3(512, 1), 256, 0, stream>>>(proj_w, 0, proj_wb, 0, (long)ATTN * DIM_, 1.f);
    convert_bf16_k<false><<<dim3(512, 1), 256, 0, stream>>>(g_w,    0, g_wb,    0, (long)DV * ATTN,  1.f);
    convert_bf16_k<false><<<dim3(512, 1), 256, 0, stream>>>(out_w,  0, out_wb,  0, (long)DIM_ * DV,  1.f);

    // xT = x^T bf16 (per batch: DIM x NPIX -> NPIX x DIM)
    transpose_convert_k<bf16_t, false><<<dim3(NPIX / 32, DIM_ / 32, B_), 256, 0, stream>>>(
        x, (long)DIM_ * NPIX, NPIX, xT, (long)NPIX * DIM_, DIM_, 1.f);

    // GEMM1: y = proj_w @ x + proj_b   (M=ATTN, N=NPIX, K=DIM)
    wmma_gemm_bf16<false><<<dim3(NPIX / 128, ATTN / 64, B_), 256, 0, stream>>>(
        proj_wb, 0, DIM_, xT, (long)NPIX * DIM_, DIM_,
        y, (long)ATTN * NPIX, NPIX, proj_b, DIM_);

    // xf = y + dwconv(y) + dwc_b
    dwconv_k<<<dim3(ATTN, B_), 256, 0, stream>>>(y, dwc_w, dwc_b, xf);

    // Staging for later GEMMs
    transpose_convert_k<bf16_t, false><<<dim3(NPIX / 32, ATTN / 32, B_), 256, 0, stream>>>(
        xf, (long)ATTN * NPIX, NPIX, xfT, (long)NPIX * ATTN, ATTN, 1.f);
    transpose_convert_k<bf16_t, true><<<dim3(NPIX / 32, DQK / 32, B_), 256, 0, stream>>>(
        xf, (long)ATTN * NPIX, NPIX, qfT, (long)NPIX * DQK, DQK, 1.f);
    transpose_convert_k<float, false><<<dim3(NPIX / 32, DV / 32, B_), 256, 0, stream>>>(
        xf + (long)2 * DQK * NPIX, (long)ATTN * NPIX, NPIX, vT, (long)NPIX * DV, DV, 1.f);
    convert_bf16_k<true><<<dim3(512, B_), 256, 0, stream>>>(
        xf + (long)DQK * NPIX, (long)ATTN * NPIX, kf_s, (long)DQK * NPIX, (long)DQK * NPIX, SCALE_);
    convert_bf16_k<false><<<dim3(512, B_), 256, 0, stream>>>(
        xf + (long)2 * DQK * NPIX, (long)ATTN * NPIX, v_s, (long)DV * NPIX, (long)DV * NPIX, SCALE_);

    // z denominator
    kmean_k<<<dim3(DQK, B_), 256, 0, stream>>>(xf, kf_mean);
    z_k<<<dim3(NPIX / 256, B_), 256, 0, stream>>>(xf, kf_mean, zbuf);

    // GEMM2: vkT = (v*s) @ (kf*s)^T   (M=DV, N=DQK, K=NPIX); Bt = kf_s as stored
    wmma_gemm_bf16<false><<<dim3(DQK / 128, DV / 64, B_), 256, 0, stream>>>(
        v_s, (long)DV * NPIX, NPIX, kf_s, (long)DQK * NPIX, NPIX,
        vkT, (long)DV * DQK, DQK, nullptr, NPIX);
    convert_bf16_k<false><<<dim3(512, B_), 256, 0, stream>>>(
        vkT, (long)DV * DQK, vkTb, (long)DV * DQK, (long)DV * DQK, 1.f);

    // GEMM3: attn^T = (vkT @ qf)^T    (M=DV, N=NPIX, K=DQK), transposed store
    wmma_gemm_bf16<true><<<dim3(NPIX / 128, DV / 64, B_), 256, 0, stream>>>(
        vkTb, (long)DV * DQK, DQK, qfT, (long)NPIX * DQK, DQK,
        attnT, (long)NPIX * DV, DV, nullptr, DQK);

    // GEMM4: gate^T = (g_w @ xf + g_b)^T  (M=DV, N=NPIX, K=ATTN), transposed
    wmma_gemm_bf16<true><<<dim3(NPIX / 128, DV / 64, B_), 256, 0, stream>>>(
        g_wb, 0, ATTN, xfT, (long)NPIX * ATTN, ATTN,
        gateT, (long)NPIX * DV, DV, g_b, ATTN);

    // Fused sigmoid-gate / z-divide / RMS-norm / +v*scale_w -> aT bf16
    epilogue_k<<<dim3(NPIX, B_), 256, 0, stream>>>(
        attnT, gateT, vT, zbuf, norm_w, scale_w, aT);

    // GEMM5: out = out_w @ a + out_b  (M=DIM, N=NPIX, K=DV) -> d_out
    wmma_gemm_bf16<false><<<dim3(NPIX / 128, DIM_ / 64, B_), 256, 0, stream>>>(
        out_wb, 0, DV, aT, (long)NPIX * DV, DV,
        out, (long)DIM_ * NPIX, NPIX, out_b, DV);
}
